// CustomMamba_74620761801336
// MI455X (gfx1250) — compile-verified, matrix operand order checked by reading, not run on hardware
//
#include <hip/hip_runtime.h>

// Problem constants (from reference setup_inputs)
#define BB 8
#define TT 24
#define NN 512
#define FF 64
#define DI 128   // 2*FF
#define DS 16
#define DR 4
#define DC 4

typedef __attribute__((ext_vector_type(16))) __bf16 bf16x16;
typedef __attribute__((ext_vector_type(8)))  float  f32x8;

// LDS layout (floats). Strides are multiples of 4 floats (rows 16B-aligned so
// ds_load_b128 vectorization is legal) and 4 mod 64 banks-wise (4m+k unique
// across the 16 rows a fragment column-read touches -> conflict-free).
#define US   68                   // sU : 32 x 64  (mix output u)
#define XZS  260                  // sXZ: 24 x 256 (xz = u @ w_in)
#define XCS  132                  // sXC: 24 x 128 (conv+silu output)
#define XDS  40                   // sXD: 24 x 36  (x_dbl)
#define YS   132                  // sY : 24 x 128 (scan output)
#define OFF_U   0
#define OFF_XZ  (OFF_U  + 32 * US)
#define OFF_XC  (OFF_XZ + TT * XZS)
#define OFF_XD  (OFF_XC + TT * XCS)
#define OFF_Y   (OFF_XD + TT * XDS)
#define SMEM_TOT (OFF_Y + TT * YS)   // 15,712 floats = 62,848 B < 64 KB

// Pre-swizzled weight fragment offsets in d_ws (bf16 elements)
#define WS_MIX  0                        // 4 ks * 4 nt * 512  = 8192
#define WS_IN   8192                     // 2 ks * 16 nt * 512 = 16384
#define WS_OUT  24576                    // 4 ks * 4 nt * 512  = 8192
#define WS_XP   32768                    // 4 ks * 3 nt * 512  = 6144
#define WS_TOTAL 38912                   // 77,824 bytes

// ---- WMMA fragment helpers (CDNA5 16x16x32 bf16 layouts) -------------------
// A 16x32: lane&15 = M row; kb = (lane>>4)*8; element e -> K = kb + e + (e>=8?8:0)
// B 32x16: lane&15 = N col; same K pattern (dense column-major B)
// C/D v8f: lane&15 = N col; VGPR j -> M = j + 8*(lane>>4)

__device__ __forceinline__ int frag_k(int kb, int e) {
    return kb + e + ((e >= 8) ? 8 : 0);
}

// A fragment built from global x/qk (concat along K): used by the mix GEMM.
// Per lane this is two contiguous 32B runs -> global_load_b128 pairs.
__device__ __forceinline__ bf16x16 load_a_mix(const float* __restrict__ x,
                                              const float* __restrict__ qk,
                                              int bi, int ni, int m0, int k0, int lane) {
    int t  = m0 + (lane & 15);
    int kb = k0 + ((lane >> 4) << 3);
    bf16x16 a;
    if (t < TT) {
        size_t base = ((((size_t)bi * TT + t) * NN) + ni) * FF;
        #pragma unroll
        for (int e = 0; e < 16; ++e) {
            int k = frag_k(kb, e);
            float v = (k < FF) ? x[base + k] : qk[base + (k - FF)];
            a[e] = (__bf16)v;
        }
    } else {
        #pragma unroll
        for (int e = 0; e < 16; ++e) a[e] = (__bf16)0.0f;
    }
    return a;
}

// A fragment from an fp32 LDS matrix (row-major, given stride), rows >= maxrow read 0.
__device__ __forceinline__ bf16x16 load_a_lds(const float* __restrict__ s, int stride,
                                              int maxrow, int m0, int k0, int lane) {
    int m  = m0 + (lane & 15);
    int kb = k0 + ((lane >> 4) << 3);
    bf16x16 a;
    #pragma unroll
    for (int e = 0; e < 16; ++e) {
        int k = frag_k(kb, e);
        a[e] = (m < maxrow) ? (__bf16)s[m * stride + k] : (__bf16)0.0f;
    }
    return a;
}

// B fragment from pre-swizzled bf16 weights: one contiguous 32B load per lane.
__device__ __forceinline__ bf16x16 load_b_ws(const __bf16* __restrict__ ws, int ntiles,
                                             int ks, int nt, int lane) {
    return *reinterpret_cast<const bf16x16*>(ws + (size_t)(((ks * ntiles + nt) * 32 + lane)) * 16);
}

__device__ __forceinline__ void store_d_lds(float* __restrict__ s, int stride, int maxrow,
                                            int m0, int n0, int lane, f32x8 c) {
    int col   = n0 + (lane & 15);
    int rbase = m0 + ((lane >> 4) << 3);
    #pragma unroll
    for (int j = 0; j < 8; ++j) {
        int row = rbase + j;
        if (row < maxrow) s[row * stride + col] = c[j];
    }
}

__device__ __forceinline__ float sigmoidf_fast(float v) {
    return 1.0f / (1.0f + __expf(-v));
}
__device__ __forceinline__ float siluf(float v) { return v * sigmoidf_fast(v); }
__device__ __forceinline__ float softplusf(float v) {
    return (v > 20.0f) ? v : log1pf(__expf(v));
}

// ---- Setup kernel: swizzle fp32 [K x N] weights into WMMA B-fragment bf16 --
// dst[((ks*ntiles + nt)*32 + lane)*16 + e] = (bf16) w[K(ks,lane,e) * ncols + N(nt,lane)]
// Columns >= nvalid are zero-padded (used for w_xproj N=36 -> 48).
__global__ __launch_bounds__(256)
void swizzle_b_kernel(const float* __restrict__ w, int ncols, int nvalid,
                      int ntiles, int ksteps, __bf16* __restrict__ dst)
{
    int idx   = blockIdx.x * blockDim.x + threadIdx.x;
    int total = ksteps * ntiles * 512;
    if (idx >= total) return;
    int e    = idx & 15;
    int lane = (idx >> 4) & 31;
    int frag = idx >> 9;
    int nt   = frag % ntiles;
    int ks   = frag / ntiles;
    int n    = nt * 16 + (lane & 15);
    int k    = ks * 32 + ((lane >> 4) << 3) + e + ((e >= 8) ? 8 : 0);
    float v  = (n < nvalid) ? w[(size_t)k * ncols + n] : 0.0f;
    dst[idx] = (__bf16)v;
}

// One workgroup (128 threads = 4 wave32) per sequence (b, n) pair.
__global__ __launch_bounds__(128)
void mamba_fused_kernel(const float* __restrict__ x,   const float* __restrict__ qk,
                        const float* __restrict__ b_mix, const float* __restrict__ conv_w,
                        const float* __restrict__ conv_b,
                        const float* __restrict__ w_dt,  const float* __restrict__ b_dt,
                        const float* __restrict__ A_log, const float* __restrict__ Dp,
                        const __bf16* __restrict__ wsMix, const __bf16* __restrict__ wsIn,
                        const __bf16* __restrict__ wsOut, const __bf16* __restrict__ wsXp,
                        float* __restrict__ out)
{
    __shared__ float sm[SMEM_TOT];
    float* sU  = sm + OFF_U;
    float* sXZ = sm + OFF_XZ;
    float* sXC = sm + OFF_XC;
    float* sXD = sm + OFF_XD;
    float* sY  = sm + OFF_Y;

    const int tid  = threadIdx.x;
    const int lane = tid & 31;
    const int wave = tid >> 5;
    const int bi   = blockIdx.x >> 9;   // / NN
    const int ni   = blockIdx.x & (NN - 1);

    // ---- Stage 1: u = [x qk] @ w_mix + b_mix  (32x128 @ 128x64) ------------
    for (int tile = wave; tile < 8; tile += 4) {
        int m0 = (tile >> 2) << 4;
        int n0 = (tile & 3) << 4;
        f32x8 c = {};
        #pragma unroll
        for (int ks = 0; ks < 4; ++ks) {
            bf16x16 a = load_a_mix(x, qk, bi, ni, m0, ks * 32, lane);
            bf16x16 b = load_b_ws(wsMix, 4, ks, tile & 3, lane);
            c = __builtin_amdgcn_wmma_f32_16x16x32_bf16(false, a, false, b,
                                                        (short)0, c, false, false);
        }
        float bm = b_mix[n0 + (lane & 15)];
        #pragma unroll
        for (int j = 0; j < 8; ++j) c[j] += bm;
        store_d_lds(sU, US, 32, m0, n0, lane, c);
    }
    __syncthreads();

    // ---- Stage 2: xz = u @ w_in  (32x64 @ 64x256) --------------------------
    for (int tile = wave; tile < 32; tile += 4) {
        int m0 = (tile >> 4) << 4;
        int n0 = (tile & 15) << 4;
        f32x8 c = {};
        #pragma unroll
        for (int ks = 0; ks < 2; ++ks) {
            bf16x16 a = load_a_lds(sU, US, 32, m0, ks * 32, lane);
            bf16x16 b = load_b_ws(wsIn, 16, ks, tile & 15, lane);
            c = __builtin_amdgcn_wmma_f32_16x16x32_bf16(false, a, false, b,
                                                        (short)0, c, false, false);
        }
        store_d_lds(sXZ, XZS, TT, m0, n0, lane, c);   // keep only t < 24
    }
    __syncthreads();

    // ---- Stage 3: causal depthwise conv (dc=4) + SiLU, per channel ----------
    {
        const int d = tid;                       // 128 threads == DI channels
        float4 cw = *reinterpret_cast<const float4*>(conv_w + d * DC);
        float cb  = conv_b[d];
        #pragma unroll 4
        for (int t = 0; t < TT; ++t) {
            float acc = cb;
            #pragma unroll
            for (int k = 0; k < DC; ++k) {
                int tp = t - (DC - 1) + k;
                float v = (tp >= 0) ? sXZ[tp * XZS + d] : 0.0f;
                acc = fmaf(v, (&cw.x)[k], acc);
            }
            sXC[t * XCS + d] = siluf(acc);
        }
    }
    __syncthreads();

    // ---- Stage 4: x_dbl = xc @ w_xproj  (24x128 @ 128x36, N padded to 48) ---
    for (int tile = wave; tile < 6; tile += 4) {
        int mt = tile / 3, nt = tile % 3;
        f32x8 c = {};
        #pragma unroll
        for (int ks = 0; ks < 4; ++ks) {
            bf16x16 a = load_a_lds(sXC, XCS, TT, mt * 16, ks * 32, lane);
            bf16x16 b = load_b_ws(wsXp, 3, ks, nt, lane);
            c = __builtin_amdgcn_wmma_f32_16x16x32_bf16(false, a, false, b,
                                                        (short)0, c, false, false);
        }
        int col   = nt * 16 + (lane & 15);
        int rbase = mt * 16 + ((lane >> 4) << 3);
        if (col < DR + 2 * DS) {
            #pragma unroll
            for (int j = 0; j < 8; ++j) {
                int row = rbase + j;
                if (row < TT) sXD[row * XDS + col] = c[j];
            }
        }
    }
    __syncthreads();

    // ---- Stage 5: selective scan, one channel per thread --------------------
    {
        const int d = tid;
        float Av[DS];
        #pragma unroll
        for (int s = 0; s < DS; ++s) Av[s] = -__expf(A_log[d * DS + s]);
        float wdt0 = w_dt[0 * DI + d], wdt1 = w_dt[1 * DI + d];
        float wdt2 = w_dt[2 * DI + d], wdt3 = w_dt[3 * DI + d];
        float bdt  = b_dt[d];
        float Dd   = Dp[d];
        float h[DS];
        #pragma unroll
        for (int s = 0; s < DS; ++s) h[s] = 0.0f;

        for (int t = 0; t < TT; ++t) {
            const float* xd = sXD + t * XDS;
            float dtr = bdt;
            dtr = fmaf(xd[0], wdt0, dtr);
            dtr = fmaf(xd[1], wdt1, dtr);
            dtr = fmaf(xd[2], wdt2, dtr);
            dtr = fmaf(xd[3], wdt3, dtr);
            float dt = softplusf(dtr);
            float ut = sXC[t * XCS + d];
            float du = dt * ut;
            float y  = 0.0f;
            #pragma unroll
            for (int s = 0; s < DS; ++s) {
                float Bv = xd[DR + s];
                float Cv = xd[DR + DS + s];
                float dA = __expf(dt * Av[s]);
                h[s] = fmaf(dA, h[s], du * Bv);
                y    = fmaf(h[s], Cv, y);
            }
            y = fmaf(ut, Dd, y);
            float z = sXZ[t * XZS + DI + d];
            y *= siluf(z);
            sY[t * YS + d] = y;
        }
    }
    __syncthreads();

    // ---- Stage 6: out = y @ w_out  (24x128 @ 128x64) -> global --------------
    for (int tile = wave; tile < 8; tile += 4) {
        int m0 = (tile >> 2) << 4;
        int n0 = (tile & 3) << 4;
        f32x8 c = {};
        #pragma unroll
        for (int ks = 0; ks < 4; ++ks) {
            bf16x16 a = load_a_lds(sY, YS, TT, m0, ks * 32, lane);
            bf16x16 b = load_b_ws(wsOut, 4, ks, tile & 3, lane);
            c = __builtin_amdgcn_wmma_f32_16x16x32_bf16(false, a, false, b,
                                                        (short)0, c, false, false);
        }
        int col   = n0 + (lane & 15);
        int rbase = m0 + ((lane >> 4) << 3);
        #pragma unroll
        for (int j = 0; j < 8; ++j) {
            int t = rbase + j;
            if (t < TT) {
                size_t o = ((((size_t)bi * TT + t) * NN) + ni) * FF + col;
                out[o] = c[j];
            }
        }
    }
}

extern "C" void kernel_launch(void* const* d_in, const int* in_sizes, int n_in,
                              void* d_out, int out_size, void* d_ws, size_t ws_size,
                              hipStream_t stream) {
    (void)in_sizes; (void)n_in; (void)ws_size; (void)out_size;
    const float* x      = (const float*)d_in[0];
    const float* qk     = (const float*)d_in[1];
    const float* w_mix  = (const float*)d_in[2];
    const float* b_mix  = (const float*)d_in[3];
    const float* w_in   = (const float*)d_in[4];
    const float* conv_w = (const float*)d_in[5];
    const float* conv_b = (const float*)d_in[6];
    const float* w_xproj= (const float*)d_in[7];
    const float* w_dt   = (const float*)d_in[8];
    const float* b_dt   = (const float*)d_in[9];
    const float* A_log  = (const float*)d_in[10];
    const float* Dp     = (const float*)d_in[11];
    const float* w_out  = (const float*)d_in[12];
    float* out = (float*)d_out;

    __bf16* ws    = (__bf16*)d_ws;
    __bf16* wsMix = ws + WS_MIX;
    __bf16* wsIn  = ws + WS_IN;
    __bf16* wsOut = ws + WS_OUT;
    __bf16* wsXp  = ws + WS_XP;

    // Pre-swizzle weights into WMMA B-fragment layout (reused by 4096 blocks).
    swizzle_b_kernel<<<(4 * 4 * 512 + 255) / 256, 256, 0, stream>>>(w_mix, 64, 64, 4, 4, wsMix);
    swizzle_b_kernel<<<(2 * 16 * 512 + 255) / 256, 256, 0, stream>>>(w_in, 256, 256, 16, 2, wsIn);
    swizzle_b_kernel<<<(4 * 4 * 512 + 255) / 256, 256, 0, stream>>>(w_out, 64, 64, 4, 4, wsOut);
    swizzle_b_kernel<<<(4 * 3 * 512 + 255) / 256, 256, 0, stream>>>(w_xproj, 36, 36, 3, 4, wsXp);

    dim3 grid(BB * NN);      // 4096 sequences, one workgroup each
    dim3 block(128);         // 4 wave32 per workgroup
    mamba_fused_kernel<<<grid, block, 0, stream>>>(x, qk, b_mix, conv_w, conv_b,
                                                   w_dt, b_dt, A_log, Dp,
                                                   wsMix, wsIn, wsOut, wsXp, out);
}
